// DCRNNEncoder_44074954392026
// MI455X (gfx1250) — matrix operand order repaired and checked by reading, not compiled, not run on hardware
//
#include <hip/hip_runtime.h>
#include <stdint.h>

// ---------------------------------------------------------------------------
// DCGRU encoder for MI455X (gfx1250, wave32). All matmuls via
// v_wmma_f32_16x16x32_bf16 (bf16 in, f32 accumulate), native bf16 converts.
// ---------------------------------------------------------------------------

typedef __attribute__((ext_vector_type(16))) __bf16 v16bf;
typedef __attribute__((ext_vector_type(8)))  __bf16 v8bf;
typedef __attribute__((ext_vector_type(8)))  float  v8f;

union Frag { v16bf v; uint4 q[2]; __bf16 b[16]; };
union H8   { v8bf  v; uint4 q;    __bf16 b[8];  };

__device__ __forceinline__ v8bf cvt8(float4 a, float4 b) {
    v8f f = {a.x, a.y, a.z, a.w, b.x, b.y, b.z, b.w};
    return __builtin_convertvector(f, v8bf);
}

// ---------------------------------------------------------------------------
// Weight prep: W [K][O] f32 (row index k = d*3+m)  ->  Wt [O][K] bf16
// ---------------------------------------------------------------------------
__global__ void wprep_kernel(const float* __restrict__ W,
                             __bf16* __restrict__ Wt,
                             int Kdim, int Odim) {
    int i = blockIdx.x * 256 + threadIdx.x;
    if (i >= Kdim * Odim) return;
    int k = i / Odim, o = i - k * Odim;
    Wt[(size_t)o * Kdim + k] = (__bf16)W[i];
}

// ---------------------------------------------------------------------------
// Diffusion kernel: builds xc[b][node][768] bf16 with K-index = feat*3 + m,
// m=0: x_cat, m=1: S@x_cat, m=2: 2*S@(S@x_cat) - x_cat.
// Grid: 32 blocks = (batch b, feature-half chunk). 256 threads = 8 waves.
// LDS: two transposed 128x128 bf16 panels (feat-major) = 64 KB.
// All three m-slices of xc are emitted from phase 2 (lane already holds the
// full triplet), so phases 0/1 touch only LDS.
// ---------------------------------------------------------------------------
__global__ __launch_bounds__(256) void diffusion_kernel(
    const float* __restrict__ sup_t,                 // [128][128] f32
    const float* __restrict__ xin_t,                 // per-t input base
    long long xinBatchStride, int xinNodeStride,
    const float* __restrict__ h,                     // [B*N][128] f32
    const float* __restrict__ gates,                 // [B*N][256] f32 (r in [:128])
    int isCand,
    __bf16* __restrict__ xc)                         // [B*N][768] bf16
{
    __shared__ __align__(16) __bf16 xT [128 * 128];  // [feat_local][node]
    __shared__ __align__(16) __bf16 x1T[128 * 128];  // [feat_local][node]

    const int tid   = threadIdx.x;
    const int b     = blockIdx.x >> 1;
    const int chunk = blockIdx.x & 1;                // 0: xin feats, 1: h feats

    // ---- phase 0: stage x_cat transposed into LDS ----
    for (int i = tid; i < 128 * 128; i += 256) {
        int node = i & 127;
        int feat = i >> 7;
        float v;
        if (chunk == 0) {
            v = xin_t[(long long)b * xinBatchStride +
                      (long long)node * xinNodeStride + feat];
        } else {
            int row = b * 128 + node;
            v = h[row * 128 + feat];
            if (isCand) v *= gates[row * 256 + feat];      // r * h
        }
        xT[feat * 128 + node] = (__bf16)v;
    }
    __syncthreads();

    const int wave = tid >> 5, lane = tid & 31;
    const int laneM = lane & 15, laneHi = lane >> 4;
    const int mt = wave;                              // node tile (8 tiles, 8 waves)
    const int nodeBase = mt * 16 + laneHi * 8;

    // A fragments of S (reused by both hops): row = node, K = graph dim
    Frag aS[4];
    #pragma unroll
    for (int kt = 0; kt < 4; ++kt) {
        int row = mt * 16 + laneM;
        int kg  = kt * 32 + laneHi * 16;
        const float4* p0 = (const float4*)(sup_t + row * 128 + kg);
        const float4* p1 = (const float4*)(sup_t + row * 128 + kg + 8);
        H8 lo, hi;
        lo.v = cvt8(p0[0], p0[1]);
        hi.v = cvt8(p1[0], p1[1]);
        aS[kt].q[0] = lo.q;
        aS[kt].q[1] = hi.q;
    }

    // ---- phase 1: x1 = S @ x_cat (this feature chunk) -> x1T (LDS only) ----
    #pragma unroll
    for (int nt = 0; nt < 8; ++nt) {
        v8f acc = {};
        int featc = nt * 16 + laneM;
        #pragma unroll
        for (int kt = 0; kt < 4; ++kt) {
            Frag bfr;
            int k0 = kt * 32 + laneHi * 16;
            const uint4* lp = (const uint4*)&xT[featc * 128 + k0];
            bfr.q[0] = lp[0]; bfr.q[1] = lp[1];
            acc = __builtin_amdgcn_wmma_f32_16x16x32_bf16(
                false, aS[kt].v, false, bfr.v, (short)0, acc, false, false);
        }
        H8 h8;
        h8.v = __builtin_convertvector(acc, v8bf);
        *(uint4*)&x1T[featc * 128 + nodeBase] = h8.q;
    }
    __syncthreads();

    // ---- phase 2: x2 = 2*(S@x1) - x_cat; emit (m0,m1,m2) triplets to xc ----
    #pragma unroll
    for (int nt = 0; nt < 8; ++nt) {
        v8f acc = {};
        int featc = nt * 16 + laneM;
        #pragma unroll
        for (int kt = 0; kt < 4; ++kt) {
            Frag bfr;
            int k0 = kt * 32 + laneHi * 16;
            const uint4* lp = (const uint4*)&x1T[featc * 128 + k0];
            bfr.q[0] = lp[0]; bfr.q[1] = lp[1];
            acc = __builtin_amdgcn_wmma_f32_16x16x32_bf16(
                false, aS[kt].v, false, bfr.v, (short)0, acc, false, false);
        }
        H8 x0, x1v;
        x0.q  = *(const uint4*)&xT [featc * 128 + nodeBase];
        x1v.q = *(const uint4*)&x1T[featc * 128 + nodeBase];
        int featG = chunk * 128 + featc;
        #pragma unroll
        for (int r = 0; r < 8; ++r) {
            float x2 = 2.0f * acc[r] - (float)x0.b[r];
            __bf16* p = xc + (size_t)(b * 128 + nodeBase + r) * 768
                           + (size_t)featG * 3;
            p[0] = x0.b[r];
            p[1] = x1v.b[r];
            p[2] = (__bf16)x2;
        }
    }
}

// ---------------------------------------------------------------------------
// GEMM:  out = xc[2048][768] @ Wt^T + bias,  fused epilogue.
// MODE 0 (NT=16): gates = sigmoid(out) -> gates buffer [row][256]
// MODE 1 (NT=8):  c = tanh(out); h' = u*h + (1-u)*c -> h and seqout
// Grid: 16 blocks x 256 threads; wave-per-16-row M tile, K-outer loop.
// ---------------------------------------------------------------------------
template <int NT, int MODE>
__global__ __launch_bounds__(256) void gemm_kernel(
    const __bf16* __restrict__ xc,
    const __bf16* __restrict__ Wt,
    const float* __restrict__ bias,
    float* __restrict__ gates,
    float* __restrict__ h,
    float* __restrict__ seqout)
{
    const int tid  = threadIdx.x;
    const int wave = tid >> 5, lane = tid & 31;
    const int laneM = lane & 15, laneHi = lane >> 4;
    const int mt   = blockIdx.x * 8 + wave;
    const int row0 = mt * 16;

    v8f acc[NT];
    v8f z = {};
    #pragma unroll
    for (int i = 0; i < NT; ++i) acc[i] = z;

    const int arow = row0 + laneM;
    for (int kt = 0; kt < 24; ++kt) {
        Frag a;
        int kb = kt * 32 + laneHi * 16;
        a.q[0] = *(const uint4*)(xc + (size_t)arow * 768 + kb);
        a.q[1] = *(const uint4*)(xc + (size_t)arow * 768 + kb + 8);
        // speculative prefetch of the next K-tile of A (global_prefetch_b8)
        __builtin_prefetch(xc + (size_t)arow * 768 + kb + 32, 0, 1);
        int k0 = kt * 32 + laneHi * 16;
        #pragma unroll
        for (int nt = 0; nt < NT; ++nt) {
            Frag bfr;
            int col = nt * 16 + laneM;
            const uint4* bp = (const uint4*)(Wt + (size_t)col * 768 + k0);
            bfr.q[0] = bp[0]; bfr.q[1] = bp[1];
            acc[nt] = __builtin_amdgcn_wmma_f32_16x16x32_bf16(
                false, a.v, false, bfr.v, (short)0, acc[nt], false, false);
        }
    }

    #pragma unroll
    for (int nt = 0; nt < NT; ++nt) {
        int col = nt * 16 + laneM;
        float bb = bias[col];
        #pragma unroll
        for (int r = 0; r < 8; ++r) {
            int row = row0 + laneHi * 8 + r;
            float v = acc[nt][r] + bb;
            if (MODE == 0) {
                // sigmoid via v_exp_f32 + v_rcp_f32
                float g = __builtin_amdgcn_rcpf(1.0f + __expf(-v));
                gates[row * 256 + col] = g;
            } else {
                // tanh(v) = 1 - 2/(exp(2v)+1)
                float e  = __expf(2.0f * v);
                float c  = 1.0f - 2.0f * __builtin_amdgcn_rcpf(e + 1.0f);
                float u  = gates[row * 256 + 128 + col];
                float hv = h[row * 128 + col];
                float hn = u * hv + (1.0f - u) * c;
                h[row * 128 + col]      = hn;
                seqout[row * 128 + col] = hn;
            }
        }
    }
}

// ---------------------------------------------------------------------------
// Host-side sequencing (graph-capture safe: only kernel launches + async d2d)
// ---------------------------------------------------------------------------
extern "C" void kernel_launch(void* const* d_in, const int* in_sizes, int n_in,
                              void* d_out, int out_size, void* d_ws, size_t ws_size,
                              hipStream_t stream) {
    (void)in_sizes; (void)n_in; (void)out_size; (void)ws_size;

    const float* inputs = (const float*)d_in[0];   // [B,T,N,D]
    const float* ihs    = (const float*)d_in[1];   // [L,B,N*H]
    const float* sup    = (const float*)d_in[2];   // [1,T,N,N]
    const float* Wg0 = (const float*)d_in[3];
    const float* bg0 = (const float*)d_in[4];
    const float* Wc0 = (const float*)d_in[5];
    const float* bc0 = (const float*)d_in[6];
    const float* Wg1 = (const float*)d_in[7];
    const float* bg1 = (const float*)d_in[8];
    const float* Wc1 = (const float*)d_in[9];
    const float* bc1 = (const float*)d_in[10];

    char* ws = (char*)d_ws;
    float*   hbuf  = (float*)(ws + 0);            // 1 MB   [B*N][128]
    float*   gates = (float*)(ws + 1048576);      // 2 MB   [B*N][256]
    float*   outs0 = (float*)(ws + 3145728);      // 32 MB  [T][B*N][128]
    __bf16*  xc    = (__bf16*)(ws + 36700160);    // 3 MB   [B*N][768]
    __bf16*  Wg0t  = (__bf16*)(ws + 39845888);
    __bf16*  Wc0t  = (__bf16*)(ws + 40239104);
    __bf16*  Wg1t  = (__bf16*)(ws + 40435712);
    __bf16*  Wc1t  = (__bf16*)(ws + 40828928);

    float* finals = (float*)d_out;                      // [L,B,N*H]
    float* outseq = (float*)d_out + (size_t)2 * 262144; // [T,B,N*H]

    wprep_kernel<<<768, 256, 0, stream>>>(Wg0, Wg0t, 768, 256);
    wprep_kernel<<<384, 256, 0, stream>>>(Wc0, Wc0t, 768, 128);
    wprep_kernel<<<768, 256, 0, stream>>>(Wg1, Wg1t, 768, 256);
    wprep_kernel<<<384, 256, 0, stream>>>(Wc1, Wc1t, 768, 128);

    for (int l = 0; l < 2; ++l) {
        hipMemcpyAsync(hbuf, ihs + (size_t)l * 262144, 262144 * sizeof(float),
                       hipMemcpyDeviceToDevice, stream);

        const float* xinBase = (l == 0) ? inputs : outs0;
        long long tStride = (l == 0) ? (long long)128 * 128       // N*D
                                     : (long long)16 * 128 * 128; // B*N*H
        long long bStride = (l == 0) ? (long long)32 * 128 * 128  // T*N*D
                                     : (long long)128 * 128;      // N*H
        const __bf16* Wgt = (l == 0) ? Wg0t : Wg1t;
        const __bf16* Wct = (l == 0) ? Wc0t : Wc1t;
        const float* bg = (l == 0) ? bg0 : bg1;
        const float* bc = (l == 0) ? bc0 : bc1;
        float* seqBase = (l == 0) ? outs0 : outseq;

        for (int t = 0; t < 32; ++t) {
            const float* sup_t = sup + (size_t)t * 128 * 128;
            const float* xin_t = xinBase + (size_t)t * tStride;

            diffusion_kernel<<<32, 256, 0, stream>>>(
                sup_t, xin_t, bStride, 128, hbuf, gates, 0, xc);
            gemm_kernel<16, 0><<<16, 256, 0, stream>>>(
                xc, Wgt, bg, gates, hbuf, nullptr);
            diffusion_kernel<<<32, 256, 0, stream>>>(
                sup_t, xin_t, bStride, 128, hbuf, gates, 1, xc);
            gemm_kernel<8, 1><<<16, 256, 0, stream>>>(
                xc, Wct, bc, gates, hbuf, seqBase + (size_t)t * 262144);
        }
        hipMemcpyAsync(finals + (size_t)l * 262144, hbuf, 262144 * sizeof(float),
                       hipMemcpyDeviceToDevice, stream);
    }
}
